// MHA_31980326486203
// MI455X (gfx1250) — matrix operand order repaired and checked by reading, not compiled, not run on hardware
//
#include <hip/hip_runtime.h>

typedef unsigned short u16;
typedef __attribute__((ext_vector_type(16))) __bf16 v16bf;
typedef __attribute__((ext_vector_type(8)))  float  v8f;
typedef __attribute__((__vector_size__(16))) int v4i_;

// ---- problem constants ----
constexpr int DIM      = 2048;
constexpr int N_HEADS  = 16;
constexpr int N_KVH    = 4;
constexpr int HD       = 128;
constexpr int Bb       = 2;
constexpr int S_       = 2048;
constexpr int MTOK     = Bb * S_;          // 4096 rows
constexpr float SCALE  = 0.08838834764831845f; // 1/sqrt(128)

// ---- workspace layout (bytes) ----
constexpr size_t OFF_XB  = 0;          // x bf16        [4096,2048]
constexpr size_t OFF_WQT = 16777216;   // Wq^T bf16     [2048,2048]
constexpr size_t OFF_WKT = 25165824;   // Wk^T bf16     [512,2048]
constexpr size_t OFF_WVT = 27262976;   // Wv^T bf16     [512,2048]
constexpr size_t OFF_WOT = 29360128;   // Wo^T bf16     [2048,2048]
constexpr size_t OFF_QB  = 37748736;   // q bf16        [B,S,16,128]
constexpr size_t OFF_KB  = 54525952;   // k bf16        [B,S,4,128]
constexpr size_t OFF_VB  = 58720256;   // v bf16        [B,S,4,128]
constexpr size_t OFF_OB  = 62914560;   // attn out bf16 [B,S,16,128]
constexpr size_t OFF_VTB = 79691776;   // v^T bf16      [B,4,128,S]

// ---- CDNA5 async global->LDS path (guarded; falls back to VGPR staging) ----
#if __has_builtin(__builtin_amdgcn_global_load_async_to_lds_b128) && \
    __has_builtin(__builtin_amdgcn_s_wait_asynccnt)
#define HAVE_ASYNC 1
#else
#define HAVE_ASYNC 0
#endif

static __device__ inline void async_ld_b128(const void* g, void* l) {
#if HAVE_ASYNC
    __builtin_amdgcn_global_load_async_to_lds_b128(
        (__attribute__((address_space(1))) v4i_*)g,
        (__attribute__((address_space(3))) v4i_*)l, 0, 0);
#else
    *reinterpret_cast<uint4*>(l) = *reinterpret_cast<const uint4*>(g);
#endif
}
static __device__ inline void wait_async0() {
#if HAVE_ASYNC
    __builtin_amdgcn_s_wait_asynccnt(0);
#endif
}

// ---- helpers ----
static __device__ inline u16 f2bf(float f) {
    unsigned int u = __builtin_bit_cast(unsigned int, f);
    u += 0x7FFFu + ((u >> 16) & 1u);           // round-to-nearest-even
    return (u16)(u >> 16);
}
static __device__ inline float bf2f(u16 h) {
    unsigned int u = ((unsigned int)h) << 16;
    return __builtin_bit_cast(float, u);
}
static __device__ inline v8f wmma_bf16f32(v16bf a, v16bf b, v8f c) {
    return __builtin_amdgcn_wmma_f32_16x16x32_bf16(false, a, false, b,
                                                   (short)0, c, false, false);
}
union FragU { uint4 u[2]; v16bf v; };

// ============================================================
// x (fp32) -> bf16, 4 elems / thread
// ============================================================
__global__ __launch_bounds__(256) void cvt_x_kernel(const float* __restrict__ in,
                                                    u16* __restrict__ out, int n4) {
    int i = blockIdx.x * blockDim.x + threadIdx.x;
    if (i >= n4) return;
    float4 v = reinterpret_cast<const float4*>(in)[i];
    unsigned long long r = (unsigned long long)f2bf(v.x)
                         | ((unsigned long long)f2bf(v.y) << 16)
                         | ((unsigned long long)f2bf(v.z) << 32)
                         | ((unsigned long long)f2bf(v.w) << 48);
    reinterpret_cast<unsigned long long*>(out)[i] = r;
}

// ============================================================
// W[K,N] fp32 -> W^T[N,K] bf16 (32x32 LDS-tiled transpose)
// ============================================================
__global__ __launch_bounds__(256) void cvt_transpose_kernel(const float* __restrict__ W,
                                                            u16* __restrict__ Wt,
                                                            int K, int N) {
    __shared__ float tile[32][33];
    const int k0 = blockIdx.y * 32, n0 = blockIdx.x * 32;
    const int c = threadIdx.x & 31, r8 = threadIdx.x >> 5;
#pragma unroll
    for (int i = 0; i < 4; ++i) {
        int r = r8 + i * 8;
        tile[r][c] = W[(size_t)(k0 + r) * N + n0 + c];
    }
    __syncthreads();
#pragma unroll
    for (int i = 0; i < 4; ++i) {
        int nr = r8 + i * 8;
        Wt[(size_t)(n0 + nr) * K + k0 + c] = f2bf(tile[c][nr]);
    }
}

// ============================================================
// V [B,S,4,128] bf16 -> V^T [B,4,128,S] bf16 (32x32 tiles)
// ============================================================
__global__ __launch_bounds__(256) void transpose_v_kernel(const u16* __restrict__ vb,
                                                          u16* __restrict__ vtb) {
    __shared__ u16 tile[32][33];
    const int s0 = blockIdx.x * 32, d0 = blockIdx.y * 32;
    const int bh = blockIdx.z, b = bh >> 2, kvh = bh & 3;
    const int c = threadIdx.x & 31, r8 = threadIdx.x >> 5;
#pragma unroll
    for (int i = 0; i < 4; ++i) {
        int r = r8 + i * 8;
        tile[r][c] = vb[(((size_t)(b * S_ + s0 + r)) * N_KVH + kvh) * HD + d0 + c];
    }
    __syncthreads();
#pragma unroll
    for (int i = 0; i < 4; ++i) {
        int nr = r8 + i * 8;
        vtb[(((size_t)(b * N_KVH + kvh)) * HD + d0 + nr) * S_ + s0 + c] = tile[c][nr];
    }
}

// ============================================================
// GEMM: C[M,N] = A[M,K] @ Bt[N,K]^T   (bf16 in, f32 acc)
// block 256 thr (8 waves), BM=128 BN=128 KT=32.
// wave-tile 64x32 -> 8 WMMA per k-step per wave.
// Double-buffered LDS, async global->LDS staging (ASYNCcnt).
// ============================================================
template <bool OUT_BF16>
__global__ __launch_bounds__(256) void gemm_bf16_kernel(const u16* __restrict__ A,
                                                        const u16* __restrict__ Bt,
                                                        void* __restrict__ Cv,
                                                        int M, int N, int K) {
    __shared__ u16 As[2][128 * 40];   // pitch 40 bf16 = 80B (16B aligned, bank-spread)
    __shared__ u16 Bs[2][128 * 40];
    const int tid  = threadIdx.x;
    const int w    = tid >> 5, lane = tid & 31;
    const int lm   = lane & 15, lh = lane >> 4;
    const int wm   = w & 1, wn = w >> 1;              // 2x4 wave grid
    const int m0   = blockIdx.y * 128, n0 = blockIdx.x * 128;

    v8f c[4][2];
#pragma unroll
    for (int i = 0; i < 4; ++i)
#pragma unroll
        for (int j = 0; j < 2; ++j)
#pragma unroll
            for (int r = 0; r < 8; ++r) c[i][j][r] = 0.f;

    const int srow = tid >> 1, scol = (tid & 1) * 16;  // staging: 128 rows x 32 cols

    auto stage = [&](int kb, int buf) {
        const u16* asrc = A  + (size_t)(m0 + srow) * K + kb + scol;
        u16* adst = &As[buf][srow * 40 + scol];
        async_ld_b128(asrc,     adst);
        async_ld_b128(asrc + 8, adst + 8);
        const u16* bsrc = Bt + (size_t)(n0 + srow) * K + kb + scol;
        u16* bdst = &Bs[buf][srow * 40 + scol];
        async_ld_b128(bsrc,     bdst);
        async_ld_b128(bsrc + 8, bdst + 8);
    };

    stage(0, 0);
    int buf = 0;
    for (int kb = 0; kb < K; kb += 32) {
        wait_async0();
        __syncthreads();                      // tile `buf` ready for everyone
        if (kb + 32 < K) stage(kb + 32, buf ^ 1);

        v16bf a[4], bf[2];
#pragma unroll
        for (int i = 0; i < 4; ++i) {         // A frag: lane=M, VGPR halves K{0-7|8-15},{16-23|24-31}
            int row = wm * 64 + i * 16 + lm;
            FragU t;
            t.u[0] = *reinterpret_cast<const uint4*>(&As[buf][row * 40 + lh * 8]);
            t.u[1] = *reinterpret_cast<const uint4*>(&As[buf][row * 40 + 16 + lh * 8]);
            a[i] = t.v;
        }
#pragma unroll
        for (int j = 0; j < 2; ++j) {         // B frag: lane=N, VGPR halves K{0-7|8-15},{16-23|24-31}
            int row = wn * 32 + j * 16 + lm;
            FragU t;
            t.u[0] = *reinterpret_cast<const uint4*>(&Bs[buf][row * 40 + lh * 8]);
            t.u[1] = *reinterpret_cast<const uint4*>(&Bs[buf][row * 40 + 16 + lh * 8]);
            bf[j] = t.v;
        }
#pragma unroll
        for (int i = 0; i < 4; ++i)
#pragma unroll
            for (int j = 0; j < 2; ++j)
                c[i][j] = wmma_bf16f32(a[i], bf[j], c[i][j]);
        buf ^= 1;
    }

    // epilogue: C layout row = r + 8*lh, col = lm
#pragma unroll
    for (int i = 0; i < 4; ++i)
#pragma unroll
        for (int j = 0; j < 2; ++j)
#pragma unroll
            for (int r = 0; r < 8; ++r) {
                int m = m0 + wm * 64 + i * 16 + r + 8 * lh;
                int n = n0 + wn * 32 + j * 16 + lm;
                if (OUT_BF16)
                    reinterpret_cast<u16*>(Cv)[(size_t)m * N + n] = f2bf(c[i][j][r]);
                else
                    reinterpret_cast<float*>(Cv)[(size_t)m * N + n] = c[i][j][r];
            }
}

// ============================================================
// In-place RoPE on bf16 [B,S,nheads,128]; one thread per pair
// ============================================================
__global__ __launch_bounds__(256) void rope_kernel(u16* __restrict__ t, int nheads) {
    int idx = blockIdx.x * blockDim.x + threadIdx.x;
    int d   = idx & 63;
    int hh  = (idx >> 6) % nheads;
    int pos = (idx / (64 * nheads)) % S_;
    int b   = idx / (64 * nheads * S_);
    size_t base = (((size_t)(b * S_ + pos)) * nheads + hh) * HD + 2 * d;
    // theta = 1000^(-2d/128) = exp(-2d * ln(1000)/128)
    float theta = __expf(-(float)(2 * d) * (6.907755278982137f / 128.0f));
    float ang = (float)pos * theta;
    float sn, cs;
    __sincosf(ang, &sn, &cs);
    float x1 = bf2f(t[base]), x2 = bf2f(t[base + 1]);
    t[base]     = f2bf(x1 * cs - x2 * sn);
    t[base + 1] = f2bf(x1 * sn + x2 * cs);
}

// ============================================================
// Flash attention (causal, GQA). block = 128 thr (4 waves).
// Each block: one (b, h, 64-query tile); each wave: 16 queries.
// kv tiles of 32; QK^T and PV both on bf16 WMMA.
// K and V^T tiles double-buffered, staged via async global->LDS.
// ============================================================
__global__ __launch_bounds__(128) void attn_kernel(const u16* __restrict__ qb,
                                                   const u16* __restrict__ kb,
                                                   const u16* __restrict__ vtb,
                                                   u16* __restrict__ ob) {
    __shared__ u16 Ks[2][32 * 136];   // K tile   [tok][d], pitch 136 (272B)
    __shared__ u16 Vt[2][128 * 40];   // V^T tile [d][tok], pitch 40  (80B)
    __shared__ u16 Ps[4 * 16 * 32];   // per-wave P scratch (wave-private)

    const int tid = threadIdx.x;
    const int w = tid >> 5, lane = tid & 31, lm = lane & 15, lh = lane >> 4;
    const int qblk = blockIdx.x, h = blockIdx.y, b = blockIdx.z;
    const int kvh  = h >> 2;                      // GQA: 4 q-heads per kv-head
    const int qpos0 = qblk * 64 + w * 16;

    // Q fragments (held in regs whole kernel): 4 chunks of K=32 over head_dim
    v16bf qf[4];
    {
        const size_t qrow = (((size_t)(b * S_ + qpos0 + lm)) * N_HEADS + h) * HD;
#pragma unroll
        for (int kc = 0; kc < 4; ++kc) {
            FragU t;
            t.u[0] = *reinterpret_cast<const uint4*>(qb + qrow + kc * 32 + lh * 8);
            t.u[1] = *reinterpret_cast<const uint4*>(qb + qrow + kc * 32 + 16 + lh * 8);
            qf[kc] = t.v;
        }
    }

    v8f o[8];
#pragma unroll
    for (int df = 0; df < 8; ++df)
#pragma unroll
        for (int r = 0; r < 8; ++r) o[df][r] = 0.f;
    float mrow[8], lrow[8];
#pragma unroll
    for (int r = 0; r < 8; ++r) { mrow[r] = -3.0e38f; lrow[r] = 0.f; }

    const int krow = tid >> 2, kseg = (tid & 3) * 32;
    auto stage = [&](int kv0, int sb) {
        // K: 32 tok x 128 d
        const u16* ksrc = kb + (((size_t)(b * S_ + kv0 + krow)) * N_KVH + kvh) * HD + kseg;
        u16* kdst = &Ks[sb][krow * 136 + kseg];
#pragma unroll
        for (int ss = 0; ss < 4; ++ss)
            async_ld_b128(ksrc + ss * 8, kdst + ss * 8);
        // V^T: 128 d x 32 tok (one row of 32 tokens per thread)
        const u16* vsrc = vtb + (((size_t)(b * N_KVH + kvh)) * HD + tid) * S_ + kv0;
        u16* vdst = &Vt[sb][tid * 40];
#pragma unroll
        for (int ss = 0; ss < 4; ++ss)
            async_ld_b128(vsrc + ss * 8, vdst + ss * 8);
    };

    const int ntiles = (qblk + 1) * 2;            // causal: kv < qblk*64 + 64
    stage(0, 0);
    int sb = 0;
    for (int t = 0; t < ntiles; ++t) {
        const int kv0 = t * 32;
        wait_async0();
        __syncthreads();                          // tile `sb` ready; prev reads done
        if (t + 1 < ntiles) stage((t + 1) * 32, sb ^ 1);

        // ---- scores: S = Q @ K^T  (16 x 32), 8 WMMA ----
        v8f sc[2];
#pragma unroll
        for (int nf = 0; nf < 2; ++nf)
#pragma unroll
            for (int r = 0; r < 8; ++r) sc[nf][r] = 0.f;
#pragma unroll
        for (int kc = 0; kc < 4; ++kc)
#pragma unroll
            for (int nf = 0; nf < 2; ++nf) {
                FragU t2;                           // B frag: lane = kv token
                const u16* base = &Ks[sb][(nf * 16 + lm) * 136 + kc * 32 + lh * 8];
                t2.u[0] = *reinterpret_cast<const uint4*>(base);
                t2.u[1] = *reinterpret_cast<const uint4*>(base + 16);
                sc[nf] = wmma_bf16f32(qf[kc], t2.v, sc[nf]);
            }

        // ---- online softmax (fp32) ----
#pragma unroll
        for (int r = 0; r < 8; ++r) {
            const int qrow_g = qpos0 + r + 8 * lh;
            float s0 = sc[0][r] * SCALE;
            float s1 = sc[1][r] * SCALE;
            if (kv0 + lm > qrow_g)       s0 = -3.0e38f;   // causal mask
            if (kv0 + 16 + lm > qrow_g)  s1 = -3.0e38f;
            float mx = fmaxf(s0, s1);
            mx = fmaxf(mx, __shfl_xor(mx, 1, 32));
            mx = fmaxf(mx, __shfl_xor(mx, 2, 32));
            mx = fmaxf(mx, __shfl_xor(mx, 4, 32));
            mx = fmaxf(mx, __shfl_xor(mx, 8, 32));
            float newm = fmaxf(mrow[r], mx);
            float scl  = __expf(mrow[r] - newm);
            float p0 = __expf(s0 - newm);
            float p1 = __expf(s1 - newm);
            float sum = p0 + p1;
            sum += __shfl_xor(sum, 1, 32);
            sum += __shfl_xor(sum, 2, 32);
            sum += __shfl_xor(sum, 4, 32);
            sum += __shfl_xor(sum, 8, 32);
            lrow[r] = lrow[r] * scl + sum;
            mrow[r] = newm;
#pragma unroll
            for (int df = 0; df < 8; ++df) o[df][r] *= scl;
            const int m = r + 8 * lh;              // C-layout row
            Ps[w * 512 + m * 32 + lm]      = f2bf(p0);
            Ps[w * 512 + m * 32 + 16 + lm] = f2bf(p1);
        }
        // Ps is wave-private: in-order DS + compiler dscnt waits order store->load

        // ---- O += P @ V  (16 x 128), 8 WMMA ----
        FragU pt;                                  // A frag of P from LDS
        pt.u[0] = *reinterpret_cast<const uint4*>(Ps + w * 512 + lm * 32 + lh * 8);
        pt.u[1] = *reinterpret_cast<const uint4*>(Ps + w * 512 + lm * 32 + 16 + lh * 8);
#pragma unroll
        for (int df = 0; df < 8; ++df) {
            FragU vt;                              // B frag: lane = output dim
            const u16* base = &Vt[sb][(df * 16 + lm) * 40 + lh * 8];
            vt.u[0] = *reinterpret_cast<const uint4*>(base);
            vt.u[1] = *reinterpret_cast<const uint4*>(base + 16);
            o[df] = wmma_bf16f32(pt.v, vt.v, o[df]);
        }
        sb ^= 1;
    }

    // ---- epilogue: normalize + store bf16 ----
#pragma unroll
    for (int df = 0; df < 8; ++df)
#pragma unroll
        for (int r = 0; r < 8; ++r) {
            const int pos = qpos0 + r + 8 * lh;
            const int d   = df * 16 + lm;
            ob[(((size_t)(b * S_ + pos)) * N_HEADS + h) * HD + d] = f2bf(o[df][r] / lrow[r]);
        }
}

// ============================================================
// launcher
// ============================================================
extern "C" void kernel_launch(void* const* d_in, const int* in_sizes, int n_in,
                              void* d_out, int out_size, void* d_ws, size_t ws_size,
                              hipStream_t stream) {
    (void)in_sizes; (void)n_in; (void)out_size; (void)ws_size;
    const float* x  = (const float*)d_in[0];
    const float* Wq = (const float*)d_in[1];
    const float* Wk = (const float*)d_in[2];
    const float* Wv = (const float*)d_in[3];
    const float* Wo = (const float*)d_in[4];
    float* out = (float*)d_out;

    char* ws = (char*)d_ws;
    u16* xb  = (u16*)(ws + OFF_XB);
    u16* Wqt = (u16*)(ws + OFF_WQT);
    u16* Wkt = (u16*)(ws + OFF_WKT);
    u16* Wvt = (u16*)(ws + OFF_WVT);
    u16* Wot = (u16*)(ws + OFF_WOT);
    u16* qb  = (u16*)(ws + OFF_QB);
    u16* kb  = (u16*)(ws + OFF_KB);
    u16* vb  = (u16*)(ws + OFF_VB);
    u16* ob  = (u16*)(ws + OFF_OB);
    u16* vtb = (u16*)(ws + OFF_VTB);

    // 1) convert inputs to bf16 (weights transposed to N-major)
    cvt_x_kernel<<<(MTOK * DIM / 4 + 255) / 256, 256, 0, stream>>>(x, xb, MTOK * DIM / 4);
    cvt_transpose_kernel<<<dim3(DIM / 32, DIM / 32), 256, 0, stream>>>(Wq, Wqt, DIM, DIM);
    cvt_transpose_kernel<<<dim3(512 / 32, DIM / 32), 256, 0, stream>>>(Wk, Wkt, DIM, 512);
    cvt_transpose_kernel<<<dim3(512 / 32, DIM / 32), 256, 0, stream>>>(Wv, Wvt, DIM, 512);
    cvt_transpose_kernel<<<dim3(DIM / 32, DIM / 32), 256, 0, stream>>>(Wo, Wot, DIM, DIM);

    // 2) QKV projections (bf16 WMMA, bf16 out)
    gemm_bf16_kernel<true><<<dim3(DIM / 128, MTOK / 128), 256, 0, stream>>>(xb, Wqt, qb, MTOK, DIM, DIM);
    gemm_bf16_kernel<true><<<dim3(512 / 128, MTOK / 128), 256, 0, stream>>>(xb, Wkt, kb, MTOK, 512, DIM);
    gemm_bf16_kernel<true><<<dim3(512 / 128, MTOK / 128), 256, 0, stream>>>(xb, Wvt, vb, MTOK, 512, DIM);

    // 3) RoPE (in-place on bf16 q, k); V -> V^T for attention B-fragments
    rope_kernel<<<Bb * S_ * N_HEADS * 64 / 256, 256, 0, stream>>>(qb, N_HEADS);
    rope_kernel<<<Bb * S_ * N_KVH * 64 / 256, 256, 0, stream>>>(kb, N_KVH);
    transpose_v_kernel<<<dim3(S_ / 32, HD / 32, Bb * N_KVH), 256, 0, stream>>>(vb, vtb);

    // 4) causal GQA flash attention
    attn_kernel<<<dim3(S_ / 64, N_HEADS, Bb), 128, 0, stream>>>(qb, kb, vtb, ob);

    // 5) output projection (fp32 out)
    gemm_bf16_kernel<false><<<dim3(DIM / 128, MTOK / 128), 256, 0, stream>>>(ob, Wot, out, MTOK, DIM, DIM);
}
